// Edge_Encoder_Residual_5239860101368
// MI455X (gfx1250) — matrix operand order repaired and checked by reading, not compiled, not run on hardware
//
#include <hip/hip_runtime.h>
#include <hip/hip_bf16.h>
#include <math.h>

// ---------------- problem constants (from reference) ----------------
#define NN    10000
#define EE    320000
#define FF    128
#define HH    8
#define CC    32
#define HC    256          // H*C
#define CONDD 128
#define ETOT  (EE + NN)    // edges + self loops

typedef __attribute__((ext_vector_type(16))) unsigned short v16u;
typedef __attribute__((ext_vector_type(8)))  unsigned short v8u;
typedef __attribute__((ext_vector_type(16))) __bf16         v16bf;
typedef __attribute__((ext_vector_type(8)))  float          v8f;

// fp32 -> bf16 round-to-nearest-even
__device__ __forceinline__ unsigned short f2bf(float f) {
    unsigned int u = __float_as_uint(f);
    unsigned int r = u + 0x7FFFu + ((u >> 16) & 1u);
    return (unsigned short)(r >> 16);
}

// float atomic max via int/uint ordering trick (init with -inf)
__device__ __forceinline__ void atomicMaxF(float* addr, float val) {
    if (val >= 0.0f) atomicMax((int*)addr, __float_as_int(val));
    else             atomicMin((unsigned int*)addr, (unsigned int)__float_as_int(val));
}

// ---------------- generic fill ----------------
__global__ void fill_kernel(float* __restrict__ p, float v, int n) {
    int i = blockIdx.x * blockDim.x + threadIdx.x;
    if (i < n) p[i] = v;
}

// ---------------- elementwise f32 -> bf16 (row-major preserved) ----------------
__global__ void cvt_bf16_kernel(const float* __restrict__ in, unsigned short* __restrict__ out, int n) {
    int i = blockIdx.x * blockDim.x + threadIdx.x;
    if (i < n) out[i] = f2bf(in[i]);
}

// ---------------- weight pack: B[K,Nc] f32 row-major -> fragment-native bf16 ----------------
// Bp[(t*Nc + n)*32 + j] = bf16(B[(32t + j)*Nc + n]),  t = K/32 tiles.
__global__ void pack_b_kernel(const float* __restrict__ B, unsigned short* __restrict__ Bp,
                              int K, int Nc) {
    int i = blockIdx.x * blockDim.x + threadIdx.x;
    if (i >= K * Nc) return;
    int k = i / Nc, n = i % Nc;
    int t = k >> 5, j = k & 31;
    Bp[((size_t)t * Nc + n) * 32 + j] = f2bf(B[i]);
}

__device__ __forceinline__ v16bf load_frag(const unsigned short* p) {
    v8u lo = *(const v8u*)(p);
    v8u hi = *(const v8u*)(p + 16);
    return __builtin_bit_cast(v16bf,
        __builtin_shufflevector(lo, hi, 0,1,2,3,4,5,6,7,8,9,10,11,12,13,14,15));
}

// ---------------- WMMA bf16 GEMM: C = A[M,K](bf16) @ Bp(packed bf16) + bias ----------------
// One wave computes a 16x64 strip: 1 A fragment shared across 4 B fragments / 4 accumulators.
// K templated (full unroll) + explicit double-buffering: next K-step's 5 fragments are
// loaded into a second register set while the current step's 4 WMMAs execute.
// grid = (Nc/64, M/16), block = 32 (wave32).
template <int K>
__global__ void wmma_gemm_bf16(const unsigned short* __restrict__ A,
                               const unsigned short* __restrict__ Bp,
                               const float* __restrict__ bias, float* __restrict__ C,
                               int Nc) {
    const int lane = threadIdx.x;        // 0..31
    const int half = lane >> 4;          // 0: lanes 0-15, 1: lanes 16-31
    const int l16  = lane & 15;
    const int nt0  = blockIdx.x * 64;    // N strip base (4 tiles of 16)
    const int mt   = blockIdx.y * 16;    // M tile base

    // A fragment base (ISA 7.12.2, 16-bit A 16x32):
    // lane<16 : elems 0..7 = K k0+0..7,   8..15 = K k0+16..23
    // lane>=16: elems 0..7 = K k0+8..15,  8..15 = K k0+24..31
    const unsigned short* arow  = A + (size_t)(mt + l16) * K + half * 8;
    // packed-B per-lane base; advance by Nc*32 elems per K-step
    const unsigned short* bbase = Bp + ((size_t)(nt0 + l16)) * 32 + half * 8;
    const size_t bstep = (size_t)Nc * 32;

    v8f acc0 = {}, acc1 = {}, acc2 = {}, acc3 = {};

    // ---- prologue: load K-step 0 ----
    v8u a0 = *(const v8u*)(arow);
    v8u a1 = *(const v8u*)(arow + 16);
    v16bf af = __builtin_bit_cast(v16bf,
        __builtin_shufflevector(a0, a1, 0,1,2,3,4,5,6,7,8,9,10,11,12,13,14,15));
    v16bf b0 = load_frag(bbase);
    v16bf b1 = load_frag(bbase + 16 * 32);
    v16bf b2 = load_frag(bbase + 32 * 32);
    v16bf b3 = load_frag(bbase + 48 * 32);

#pragma unroll
    for (int k0 = 0; k0 < K; k0 += 32) {
        v16bf naf, nb0, nb1, nb2, nb3;
        if (k0 + 32 < K) {
            // ---- load next K-step into the second register set ----
            const unsigned short* arp = arow + k0 + 32;
            const unsigned short* bp  = bbase + (size_t)((k0 >> 5) + 1) * bstep;
            if (k0 + 64 < K) {                    // prefetch one more step ahead
                __builtin_prefetch(arp + 32, 0, 3);
                __builtin_prefetch(bp + bstep, 0, 3);
            }
            v8u na0 = *(const v8u*)(arp);
            v8u na1 = *(const v8u*)(arp + 16);
            naf = __builtin_bit_cast(v16bf,
                __builtin_shufflevector(na0, na1, 0,1,2,3,4,5,6,7,8,9,10,11,12,13,14,15));
            nb0 = load_frag(bp);
            nb1 = load_frag(bp + 16 * 32);
            nb2 = load_frag(bp + 32 * 32);
            nb3 = load_frag(bp + 48 * 32);
        }
        // ---- current step's WMMAs (overlap with next step's in-flight loads) ----
        acc0 = __builtin_amdgcn_wmma_f32_16x16x32_bf16(false, af, false, b0, (short)0, acc0, false, false);
        acc1 = __builtin_amdgcn_wmma_f32_16x16x32_bf16(false, af, false, b1, (short)0, acc1, false, false);
        acc2 = __builtin_amdgcn_wmma_f32_16x16x32_bf16(false, af, false, b2, (short)0, acc2, false, false);
        acc3 = __builtin_amdgcn_wmma_f32_16x16x32_bf16(false, af, false, b3, (short)0, acc3, false, false);
        if (k0 + 32 < K) {
            af = naf; b0 = nb0; b1 = nb1; b2 = nb2; b3 = nb3;
        }
    }

    // ---- store (f32 C/D 16x16: lane<16 rows mt+0..7, lane>=16 rows mt+8..15; col = l16)
#pragma unroll
    for (int nb = 0; nb < 4; ++nb) {
        const v8f* ap = (nb == 0) ? &acc0 : (nb == 1) ? &acc1 : (nb == 2) ? &acc2 : &acc3;
        int col = nt0 + nb * 16 + l16;
        float b = bias ? bias[col] : 0.0f;
#pragma unroll
        for (int r = 0; r < 8; ++r) {
            int row = mt + r + half * 8;
            C[(size_t)row * Nc + col] = (*ap)[r] + b;
        }
    }
}

// ---------------- edge_attr sum (for self-loop mean fill) ----------------
__global__ void ea_sum_kernel(const float* __restrict__ ea, float* __restrict__ out_sum) {
    __shared__ float sh[256];
    int tid = threadIdx.x;
    float s = 0.0f;
    for (int i = blockIdx.x * 256 + tid; i < EE; i += gridDim.x * 256) s += ea[i];
    sh[tid] = s;
    __syncthreads();
    for (int off = 128; off > 0; off >>= 1) {
        if (tid < off) sh[tid] += sh[tid + off];
        __syncthreads();
    }
    if (tid == 0) atomicAdd(out_sum, sh[0]);
}

// ---------------- residual GEMV partials: res_acc[j] += sum_e ea[e]*W[e,j] ----------------
__global__ void res_partial_kernel(const float* __restrict__ ea, const float* __restrict__ W,
                                   float* __restrict__ res_acc) {
    const int j  = threadIdx.x;                 // output column 0..255
    const int e0 = blockIdx.x * 625;
    float s = 0.0f;
    int eend = e0 + 625; if (eend > EE) eend = EE;
    for (int e = e0; e < eend; ++e) s += ea[e] * W[(size_t)e * HC + j];
    atomicAdd(&res_acc[j], s);
}

// res[j] = relu(res_acc[j] + res_b[j]);   ea_mean = ea_sum / E
__global__ void res_final_kernel(const float* __restrict__ res_acc, const float* __restrict__ res_b,
                                 float* __restrict__ res, const float* __restrict__ ea_sum,
                                 float* __restrict__ ea_mean) {
    int j = threadIdx.x;
    if (j < HC) res[j] = fmaxf(res_acc[j] + res_b[j], 0.0f);
    if (j == 0) ea_mean[0] = ea_sum[0] / (float)EE;
}

// ---------------- pass 1: per-(edge,head) logit + atomic max over dst ----------------
__global__ void edge_logits_kernel(const int* __restrict__ eidx, const float* __restrict__ ea,
                                   const float* __restrict__ ea_mean,
                                   const float* __restrict__ xl, const float* __restrict__ xr,
                                   const float* __restrict__ We, const float* __restrict__ att,
                                   float* __restrict__ logits, float* __restrict__ m) {
    int idx = blockIdx.x * blockDim.x + threadIdx.x;
    if (idx >= ETOT * HH) return;
    int e = idx / HH, h = idx % HH;
    int src, dst; float eav;
    if (e < EE) { src = eidx[e]; dst = eidx[EE + e]; eav = ea[e]; }
    else        { src = e - EE;  dst = e - EE;       eav = ea_mean[0]; }
    const float* xlp  = xl  + (size_t)src * HC + h * CC;
    const float* xrp  = xr  + (size_t)dst * HC + h * CC;
    const float* wep  = We  + h * CC;
    const float* attp = att + h * CC;
    float a = 0.0f;
#pragma unroll 8
    for (int c = 0; c < CC; ++c) {
        float v = xlp[c] + xrp[c] + eav * wep[c];
        v = (v > 0.0f) ? v : 0.2f * v;           // leaky_relu(0.2)
        a += attp[c] * v;
    }
    logits[idx] = a;
    atomicMaxF(&m[dst * HH + h], a);
}

// ---------------- pass 2: exp(a - m[dst]) + atomic sum ----------------
__global__ void edge_exp_kernel(const int* __restrict__ eidx,
                                const float* __restrict__ m, float* __restrict__ logits,
                                float* __restrict__ s) {
    int idx = blockIdx.x * blockDim.x + threadIdx.x;
    if (idx >= ETOT * HH) return;
    int e = idx / HH, h = idx % HH;
    int dst = (e < EE) ? eidx[EE + e] : (e - EE);
    float ex = __expf(logits[idx] - m[dst * HH + h]);
    logits[idx] = ex;
    atomicAdd(&s[dst * HH + h], ex);
}

// ---------------- pass 3: acc[dst] += xl[src] * alpha ----------------
__global__ void edge_scatter_kernel(const int* __restrict__ eidx,
                                    const float* __restrict__ xl, const float* __restrict__ logits,
                                    const float* __restrict__ s, float* __restrict__ acc) {
    int idx = blockIdx.x * blockDim.x + threadIdx.x;
    if (idx >= ETOT * HH) return;
    int e = idx / HH, h = idx % HH;
    int src, dst;
    if (e < EE) { src = eidx[e]; dst = eidx[EE + e]; }
    else        { src = e - EE;  dst = e - EE; }
    float alpha = logits[idx] / (s[dst * HH + h] + 1e-16f);
    const float* xlp = xl + (size_t)src * HC + h * CC;
    float* ap = acc + (size_t)dst * HC + h * CC;
#pragma unroll 4
    for (int c = 0; c < CC; ++c) atomicAdd(&ap[c], xlp[c] * alpha);
}

// ---------------- hbf = bf16(relu(acc + bias) + res)  (next layer's A matrix) ----------------
__global__ void finalize_kernel(const float* __restrict__ acc, const float* __restrict__ bias,
                                const float* __restrict__ res, unsigned short* __restrict__ hbf) {
    int i = blockIdx.x * blockDim.x + threadIdx.x;
    if (i >= NN * HC) return;
    int j = i & (HC - 1);
    hbf[i] = f2bf(fmaxf(acc[i] + bias[j], 0.0f) + res[j]);
}

// =============================================================================
extern "C" void kernel_launch(void* const* d_in, const int* in_sizes, int n_in,
                              void* d_out, int out_size, void* d_ws, size_t ws_size,
                              hipStream_t stream) {
    (void)in_sizes; (void)n_in; (void)out_size; (void)ws_size;
    const float* x      = (const float*)d_in[0];
    const float* eattr  = (const float*)d_in[1];   // [E]
    const float* Wl0    = (const float*)d_in[2];
    const float* bl0    = (const float*)d_in[3];
    const float* Wr0    = (const float*)d_in[4];
    const float* br0    = (const float*)d_in[5];
    const float* We0    = (const float*)d_in[6];
    const float* att0   = (const float*)d_in[7];
    const float* bias0  = (const float*)d_in[8];
    const float* Wl_r   = (const float*)d_in[9];
    const float* bl_r   = (const float*)d_in[10];
    const float* Wr_r   = (const float*)d_in[11];
    const float* br_r   = (const float*)d_in[12];
    const float* We_r   = (const float*)d_in[13];
    const float* att_r  = (const float*)d_in[14];
    const float* bias_r = (const float*)d_in[15];
    const float* res_W  = (const float*)d_in[16];
    const float* res_b  = (const float*)d_in[17];
    const float* hist_W = (const float*)d_in[18];
    const float* hist_b = (const float*)d_in[19];
    const int*   eidx   = (const int*)d_in[20];    // [2,E]
    float* out = (float*)d_out;                    // [1,N,COND]

    // ---- workspace carving ----
    float* ws = (float*)d_ws;
    size_t o = 0;
    float* xl      = ws + o; o += (size_t)NN * HC;
    float* xr      = ws + o; o += (size_t)NN * HC;
    float* acc     = ws + o; o += (size_t)NN * HC;
    float* logits  = ws + o; o += (size_t)ETOT * HH;
    float* mbuf    = ws + o; o += (size_t)NN * HH;
    float* sbuf    = ws + o; o += (size_t)NN * HH;
    float* res_acc = ws + o; o += HC;
    float* res     = ws + o; o += HC;
    float* ea_sum  = ws + o; o += 1;
    float* ea_mean = ws + o; o += 3;               // +pad to keep ushort region 4B aligned
    unsigned short* us = (unsigned short*)(ws + o);
    size_t uo = 0;
    unsigned short* xbf    = us + uo; uo += (size_t)NN * FF;       // bf16 x
    unsigned short* hbf    = us + uo; uo += (size_t)NN * HC;       // bf16 h
    unsigned short* Wl0p   = us + uo; uo += (size_t)FF * HC;
    unsigned short* Wr0p   = us + uo; uo += (size_t)FF * HC;
    unsigned short* Wlrp   = us + uo; uo += (size_t)2 * HC * HC;
    unsigned short* Wrrp   = us + uo; uo += (size_t)2 * HC * HC;
    unsigned short* histWp = us + uo; uo += (size_t)HC * CONDD;

    const int EHT = ETOT * HH;
    const dim3 eblk(256), egrd((EHT + 255) / 256);
    const float NEG_INF = -__builtin_huge_valf();

    // ---- one-shot conversions / weight packing ----
    cvt_bf16_kernel<<<(NN * FF + 255) / 256, 256, 0, stream>>>(x, xbf, NN * FF);
    pack_b_kernel<<<(FF * HC + 255) / 256, 256, 0, stream>>>(Wl0, Wl0p, FF, HC);
    pack_b_kernel<<<(FF * HC + 255) / 256, 256, 0, stream>>>(Wr0, Wr0p, FF, HC);
    for (int i = 0; i < 2; ++i) {
        pack_b_kernel<<<(HC * HC + 255) / 256, 256, 0, stream>>>(Wl_r + (size_t)i * HC * HC,
                                                                 Wlrp + (size_t)i * HC * HC, HC, HC);
        pack_b_kernel<<<(HC * HC + 255) / 256, 256, 0, stream>>>(Wr_r + (size_t)i * HC * HC,
                                                                 Wrrp + (size_t)i * HC * HC, HC, HC);
    }
    pack_b_kernel<<<(HC * CONDD + 255) / 256, 256, 0, stream>>>(hist_W, histWp, HC, CONDD);

    // ---- residual GEMV + edge-attr mean ----
    fill_kernel<<<1, 256, 0, stream>>>(res_acc, 0.0f, HC);
    fill_kernel<<<1, 32, 0, stream>>>(ea_sum, 0.0f, 1);
    ea_sum_kernel<<<256, 256, 0, stream>>>(eattr, ea_sum);
    res_partial_kernel<<<512, 256, 0, stream>>>(eattr, res_W, res_acc);   // 512*625 = E
    res_final_kernel<<<1, 256, 0, stream>>>(res_acc, res_b, res, ea_sum, ea_mean);

    // ---- 3 GATv2 layers ----
    for (int layer = 0; layer < 3; ++layer) {
        const float *bl, *br, *We, *att, *bias;
        const unsigned short *Ain, *Wlp_, *Wrp_;
        if (layer == 0) {
            Wlp_ = Wl0p; bl = bl0; Wrp_ = Wr0p; br = br0;
            We = We0; att = att0; bias = bias0;
            Ain = xbf;
        } else {
            int i = layer - 1;
            Wlp_ = Wlrp + (size_t)i * HC * HC; bl = bl_r + (size_t)i * HC;
            Wrp_ = Wrrp + (size_t)i * HC * HC; br = br_r + (size_t)i * HC;
            We = We_r + (size_t)i * HC;        att = att_r + (size_t)i * HC;
            bias = bias_r + (size_t)i * HC;
            Ain = hbf;
        }
        // node transforms via WMMA: 16x64 strip per wave
        if (layer == 0) {
            wmma_gemm_bf16<FF><<<dim3(HC / 64, NN / 16), 32, 0, stream>>>(Ain, Wlp_, bl, xl, HC);
            wmma_gemm_bf16<FF><<<dim3(HC / 64, NN / 16), 32, 0, stream>>>(Ain, Wrp_, br, xr, HC);
        } else {
            wmma_gemm_bf16<HC><<<dim3(HC / 64, NN / 16), 32, 0, stream>>>(Ain, Wlp_, bl, xl, HC);
            wmma_gemm_bf16<HC><<<dim3(HC / 64, NN / 16), 32, 0, stream>>>(Ain, Wrp_, br, xr, HC);
        }
        // softmax state init
        fill_kernel<<<(NN * HH + 255) / 256, 256, 0, stream>>>(mbuf, NEG_INF, NN * HH);
        fill_kernel<<<(NN * HH + 255) / 256, 256, 0, stream>>>(sbuf, 0.0f, NN * HH);
        fill_kernel<<<(NN * HC + 255) / 256, 256, 0, stream>>>(acc, 0.0f, NN * HC);
        // edge passes
        edge_logits_kernel<<<egrd, eblk, 0, stream>>>(eidx, eattr, ea_mean, xl, xr, We, att, logits, mbuf);
        edge_exp_kernel<<<egrd, eblk, 0, stream>>>(eidx, mbuf, logits, sbuf);
        edge_scatter_kernel<<<egrd, eblk, 0, stream>>>(eidx, xl, logits, sbuf, acc);
        // h(bf16) = relu(acc + bias) + res
        finalize_kernel<<<(NN * HC + 255) / 256, 256, 0, stream>>>(acc, bias, res, hbf);
    }

    // ---- condition head: out = h @ hist_W + hist_b  (WMMA) ----
    wmma_gemm_bf16<HC><<<dim3(CONDD / 64, NN / 16), 32, 0, stream>>>(hbf, histWp, hist_b, out, CONDD);
}
